// ExpanderLinear_17849884082830
// MI455X (gfx1250) — matrix operand order repaired and checked by reading, not compiled, not run on hardware
//
#include <hip/hip_runtime.h>

// out[8192,4096] = x[8192,4096] @ (W*mask)[4096,4096]^T
// Dense GEMM, bf16 inputs (fused mask), f32 WMMA accumulation.

#define D_IN   4096
#define D_OUT  4096
#define BATCH  8192

#define BM 128
#define BN 128
#define BK 32
#define NTHREADS 256
#define KTILES (D_IN / BK)          // 128
#define LDS_ROW 20                  // padded row stride in dwords (32 bf16 = 16 dwords + 4 pad)

typedef __attribute__((ext_vector_type(16))) __bf16        v16bf;
typedef __attribute__((ext_vector_type(8)))  float         v8f;
typedef __attribute__((ext_vector_type(8)))  unsigned int  v8u;

// Hardware packed f32->bf16 conversion (RNE): a -> [15:0], b -> [31:16].
// The clang builtin is not declared for gfx1250 on this toolchain, so emit
// the VOP3 instruction directly.
__device__ __forceinline__ unsigned pack_bf16(float a, float b) {
  unsigned r;
  asm volatile("v_cvt_pk_bf16_f32 %0, %1, %2" : "=v"(r) : "v"(a), "v"(b));
  return r;
}

__global__ __launch_bounds__(NTHREADS)
void expander_gemm_bf16(const float* __restrict__ x,
                        const float* __restrict__ w,
                        const float* __restrict__ mask,
                        float* __restrict__ out) {
  // [buf][A=0/B=1][row][dword]; 2*2*128*20*4 = 40 KB
  __shared__ __align__(16) unsigned lds[2][2][BM][LDS_ROW];

  const int tid  = threadIdx.x;
  const int lane = tid & 31;
  const int wave = tid >> 5;
  const int wm   = wave >> 2;          // 0..1  (64-row slab)
  const int wn   = wave & 3;           // 0..3  (32-col slab)
  const int hsel = (lane < 16) ? 0 : 1;
  const int lrow = lane & 15;

  // ---- block swizzle: group 16 M-tiles per N sweep (L2 reuse) ----
  const int num_pid_m = BATCH / BM;    // 64
  const int num_pid_n = D_OUT / BN;    // 32
  const int GROUP_M   = 16;
  int pid        = blockIdx.x;
  int group_size = GROUP_M * num_pid_n;
  int gid        = pid / group_size;
  int first_m    = gid * GROUP_M;
  int rem        = num_pid_m - first_m;
  if (rem > GROUP_M) rem = GROUP_M;
  int pid_m = first_m + (pid % group_size) % rem;
  int pid_n = (pid % group_size) / rem;

  const int m0 = pid_m * BM;
  const int n0 = pid_n * BN;

  // ---- staging: thread t covers row (t>>1), k-chunk (t&1)*16 floats ----
  const int srow  = tid >> 1;
  const int skoff = (tid & 1) * 16;              // float offset within BK
  const float* aptr = x    + (size_t)(m0 + srow) * D_IN + skoff;
  const float* wptr = w    + (size_t)(n0 + srow) * D_IN + skoff;
  const float* mptr = mask + (size_t)(n0 + srow) * D_IN + skoff;

  unsigned apk[8], bpk[8];

  auto stage = [&](int kt) {
    const float4* a4 = (const float4*)(aptr + (size_t)kt * BK);
    const float4* w4 = (const float4*)(wptr + (size_t)kt * BK);
    const float4* m4 = (const float4*)(mptr + (size_t)kt * BK);
#pragma unroll
    for (int i = 0; i < 4; ++i) {
      float4 av = a4[i];
      float4 wv = w4[i];
      float4 mv = m4[i];
      apk[2*i+0] = pack_bf16(av.x, av.y);
      apk[2*i+1] = pack_bf16(av.z, av.w);
      bpk[2*i+0] = pack_bf16(wv.x * mv.x, wv.y * mv.y);
      bpk[2*i+1] = pack_bf16(wv.z * mv.z, wv.w * mv.w);
    }
  };

  auto commit = [&](int buf) {
    unsigned* al = &lds[buf][0][srow][skoff >> 1];   // dword col 0 or 8
    unsigned* bl = &lds[buf][1][srow][skoff >> 1];
    ((uint4*)al)[0] = make_uint4(apk[0], apk[1], apk[2], apk[3]);
    ((uint4*)al)[1] = make_uint4(apk[4], apk[5], apk[6], apk[7]);
    ((uint4*)bl)[0] = make_uint4(bpk[0], bpk[1], bpk[2], bpk[3]);
    ((uint4*)bl)[1] = make_uint4(bpk[4], bpk[5], bpk[6], bpk[7]);
  };

  // A fragment (16x32 bf16): lane<16 -> K chunks [0..7],[16..23]; lane>=16 -> [8..15],[24..31]
  auto loadA = [&](int buf, int i) -> v16bf {
    const unsigned* p = &lds[buf][0][wm * 64 + i * 16 + lrow][hsel * 4];
    uint4 lo = *(const uint4*)(p);
    uint4 hi = *(const uint4*)(p + 8);
    v8u v;
    v[0]=lo.x; v[1]=lo.y; v[2]=lo.z; v[3]=lo.w;
    v[4]=hi.x; v[5]=hi.y; v[6]=hi.z; v[7]=hi.w;
    return __builtin_bit_cast(v16bf, v);
  };

  // B fragment (32x16 bf16): lane = column n; lane<16 -> K chunks [0..7],[16..23]; lane>=16 -> [8..15],[24..31]
  auto loadB = [&](int buf, int j) -> v16bf {
    const unsigned* p = &lds[buf][1][wn * 32 + j * 16 + lrow][hsel * 4];
    uint4 lo = *(const uint4*)(p);
    uint4 hi = *(const uint4*)(p + 8);
    v8u v;
    v[0]=lo.x; v[1]=lo.y; v[2]=lo.z; v[3]=lo.w;
    v[4]=hi.x; v[5]=hi.y; v[6]=hi.z; v[7]=hi.w;
    return __builtin_bit_cast(v16bf, v);
  };

  v8f acc[4][2] = {};

  stage(0);
  commit(0);
  __syncthreads();

  for (int kt = 0; kt < KTILES; ++kt) {
    const int buf = kt & 1;
    if (kt + 1 < KTILES) stage(kt + 1);   // global loads hide under WMMA

    v16bf af[4], bfrag[2];
#pragma unroll
    for (int i = 0; i < 4; ++i) af[i] = loadA(buf, i);
#pragma unroll
    for (int j = 0; j < 2; ++j) bfrag[j] = loadB(buf, j);

#pragma unroll
    for (int i = 0; i < 4; ++i)
#pragma unroll
      for (int j = 0; j < 2; ++j)
        acc[i][j] = __builtin_amdgcn_wmma_f32_16x16x32_bf16(
            false, af[i], false, bfrag[j], (short)0, acc[i][j], false, false);

    if (kt + 1 < KTILES) commit(buf ^ 1);
    __syncthreads();
  }

  // ---- epilogue: C/D layout: VGPR r -> M = r + (lane<16 ? 0 : 8), N = lane%16 ----
#pragma unroll
  for (int i = 0; i < 4; ++i) {
    const int rbase = m0 + wm * 64 + i * 16 + hsel * 8;
#pragma unroll
    for (int j = 0; j < 2; ++j) {
      const int col = n0 + wn * 32 + j * 16 + lrow;
#pragma unroll
      for (int r = 0; r < 8; ++r) {
        out[(size_t)(rbase + r) * D_OUT + col] = acc[i][j][r];
      }
    }
  }
}

extern "C" void kernel_launch(void* const* d_in, const int* in_sizes, int n_in,
                              void* d_out, int out_size, void* d_ws, size_t ws_size,
                              hipStream_t stream) {
  (void)in_sizes; (void)n_in; (void)out_size; (void)d_ws; (void)ws_size;
  const float* x    = (const float*)d_in[0];
  const float* w    = (const float*)d_in[1];
  const float* mask = (const float*)d_in[2];
  float* out        = (float*)d_out;

  const int num_blocks = (BATCH / BM) * (D_OUT / BN);   // 64 * 32 = 2048
  expander_gemm_bf16<<<num_blocks, NTHREADS, 0, stream>>>(x, w, mask, out);
}
